// MHA_89034672046283
// MI455X (gfx1250) — matrix operand order repaired and checked by reading, not compiled, not run on hardware
//
#include <hip/hip_runtime.h>
#include <hip/hip_bf16.h>

#define D_MODEL   1024
#define NUM_HEADS 16
#define SEQ_LEN   2048
#define BATCH     2
#define DK        64        // head dim

typedef __attribute__((ext_vector_type(16))) _Float16 v16h;
typedef __attribute__((ext_vector_type(8)))  _Float16 v8h;
typedef __attribute__((ext_vector_type(8)))  float    v8f;
typedef __attribute__((ext_vector_type(4)))  unsigned int v4u;
typedef __attribute__((ext_vector_type(8)))  int      v8i;
typedef __attribute__((ext_vector_type(4)))  int      v4i;

#if defined(__has_builtin)
#if __has_builtin(__builtin_amdgcn_tensor_load_to_lds)
#define HAVE_TDM 1
#endif
#endif
#ifndef HAVE_TDM
#define HAVE_TDM 0
#endif

// ---------------------------------------------------------------------------
// WMMA wrapper: D = A(16x32 f16) * B(32x16 f16) + C(16x16 f32)
// ---------------------------------------------------------------------------
__device__ __forceinline__ v8f wmma_f16(v16h a, v16h b, v8f c) {
  return __builtin_amdgcn_wmma_f32_16x16x32_f16(
      /*neg_a=*/false, a, /*neg_b=*/false, b,
      /*c_mod=*/(short)0, c, /*reuse_a=*/false, /*reuse_b=*/false);
}

// A fragment (16x32, f16), contiguous K order matching B:
// halves 0..15 -> K = k0+2*koff..+15,
// koff = (lane&16)>>1. Works for global or LDS source pointers.
__device__ __forceinline__ v16h load_a_frag(const _Float16* __restrict__ row,
                                            int k0, int koff) {
  v8h g0 = *(const v8h*)(row + k0 + 2 * koff);
  v8h g1 = *(const v8h*)(row + k0 + 2 * koff + 8);
  return __builtin_shufflevector(g0, g1, 0, 1, 2, 3, 4, 5, 6, 7,
                                         8, 9, 10, 11, 12, 13, 14, 15);
}

// ---------------------------------------------------------------------------
// TDM: 2D tile load global -> LDS (descriptor per CDNA5 ISA ch.8)
// ---------------------------------------------------------------------------
#if HAVE_TDM
__device__ __forceinline__ void tdm_load_2d(_Float16* lds_dst, const _Float16* gsrc,
                                            unsigned tensor_d0, unsigned tensor_d1,
                                            unsigned tile_d0, unsigned tile_d1,
                                            unsigned stride0 /* halves */) {
  unsigned long long ga = (unsigned long long)(uintptr_t)gsrc;
  unsigned lds_addr = (unsigned)(uintptr_t)lds_dst;  // LDS offset (low 32b of as(3))
  v4u g0;
  g0[0] = 1u;                                              // count=1, user desc
  g0[1] = lds_addr;                                        // [63:32] lds_addr
  g0[2] = (unsigned)ga;                                    // [95:64] global[31:0]
  g0[3] = (unsigned)((ga >> 32) & 0x01FFFFFFu) | (2u << 30); // global[56:32], type=2
  v8i g1;
  g1[0] = (int)(1u << 16);                                 // data_size=1 (2 bytes)
  g1[1] = (int)((tensor_d0 & 0xFFFFu) << 16);              // tensor_dim0[15:0]
  g1[2] = (int)((tensor_d0 >> 16) | ((tensor_d1 & 0xFFFFu) << 16));
  g1[3] = (int)((tensor_d1 >> 16) | (tile_d0 << 16));      // tile_dim0
  g1[4] = (int)(tile_d1 & 0xFFFFu);                        // tile_dim1, tile_dim2=0
  g1[5] = (int)stride0;                                    // tensor_dim0_stride
  g1[6] = 0;
  g1[7] = 0;
  v4i z4 = {0, 0, 0, 0};
#if __clang_major__ >= 23
  v8i z8 = {0, 0, 0, 0, 0, 0, 0, 0};
  __builtin_amdgcn_tensor_load_to_lds(g0, g1, z4, z4, z8, 0);
#else
  __builtin_amdgcn_tensor_load_to_lds(g0, g1, z4, z4, 0);
#endif
}
#else
// Fallback: cooperative copy by `nthr` threads (rows x rowlen halves).
__device__ __forceinline__ void coop_copy(_Float16* __restrict__ dst,
                                          const _Float16* __restrict__ src,
                                          int rows, int rowlen, int gstride,
                                          int tid, int nthr) {
  int chunks = rows * (rowlen >> 3);
  for (int i = tid; i < chunks; i += nthr) {
    int r = i / (rowlen >> 3);
    int c = (i - r * (rowlen >> 3)) * 8;
    *(v8h*)(dst + r * rowlen + c) = *(const v8h*)(src + (size_t)r * gstride + c);
  }
}
#endif

// ---------------------------------------------------------------------------
// f32 -> f16 elementwise convert
// ---------------------------------------------------------------------------
__global__ void cvt_f32_to_f16_kernel(const float* __restrict__ in,
                                      _Float16* __restrict__ out, int n) {
  int i = blockIdx.x * blockDim.x + threadIdx.x;
  if (i < n) out[i] = (_Float16)in[i];
}

// ---------------------------------------------------------------------------
// GEMM: Y[m, o] = sum_i A[m, i] * W[o, i]   (A @ W^T, einsum 'mi,oi->mo')
// 4-wave block -> 64(M) x 128(N) tile; K staged in 64-wide chunks in
// double-buffered LDS by the TDM (wave 0 issues; TENSORcnt pipelined).
// Each wave: 32x64 quadrant; all frags loaded before the 16 WMMAs so the
// DS loads clause together and latency hides under the matrix pipe.
// ---------------------------------------------------------------------------
template <bool F32OUT>
__global__ __launch_bounds__(128)
void gemm_awT_kernel(const _Float16* __restrict__ A,
                     const _Float16* __restrict__ W,
                     void* __restrict__ out, int M, int N, int K) {
  __shared__ _Float16 Al[2][64 * 64];    // 8 KB x2
  __shared__ _Float16 Bl[2][128 * 64];   // 16 KB x2

  const int tid  = threadIdx.x;
  const int lane = tid & 31;
  const int wid  = tid >> 5;
  const int wm   = (wid & 1) * 32;       // wave's M offset in block tile
  const int wn   = (wid >> 1) * 64;      // wave's N offset in block tile
  const int m0 = blockIdx.x * 64;
  const int n0 = blockIdx.y * 128;

  const int mlane = lane & 15;
  const int koff  = (lane & 16) >> 1;
  const int nlane = lane & 15;
  const int kb    = lane & 16;

  v8f acc[2][4] = {};
  int buf = 0;

#if HAVE_TDM
  if (tid < 32) {
    tdm_load_2d(&Al[0][0], A + (size_t)m0 * K, K, M, 64, 64, K);
    tdm_load_2d(&Bl[0][0], W + (size_t)n0 * K, K, N, 64, 128, K);
  }
#endif
  for (int k0 = 0; k0 < K; k0 += 64) {
    const bool more = (k0 + 64 < K);
    __syncthreads();  // all readers of buf^1 (prev chunk) have drained
#if HAVE_TDM
    if (tid < 32) {
      if (more) {
        tdm_load_2d(&Al[buf ^ 1][0], A + (size_t)m0 * K + k0 + 64, K, M, 64, 64, K);
        tdm_load_2d(&Bl[buf ^ 1][0], W + (size_t)n0 * K + k0 + 64, K, N, 64, 128, K);
        __builtin_amdgcn_s_wait_tensorcnt(2);  // in-order: current chunk done
      } else {
        __builtin_amdgcn_s_wait_tensorcnt(0);
      }
    }
#else
    coop_copy(&Al[buf][0], A + (size_t)m0 * K + k0, 64, 64, K, tid, 128);
    coop_copy(&Bl[buf][0], W + (size_t)n0 * K + k0, 128, 64, K, tid, 128);
#endif
    __syncthreads();  // publish current chunk
    const _Float16* Ab = &Al[buf][0];
    const _Float16* Bb = &Bl[buf][0];

    // load all fragments for the 64-wide K chunk first
    v16h af[2][2];   // [ks][m-half]
    v16h bf[2][4];   // [ks][n-tile]
#pragma unroll
    for (int s = 0; s < 2; ++s) {
      af[s][0] = load_a_frag(Ab + (wm + mlane) * 64, s * 32, koff);
      af[s][1] = load_a_frag(Ab + (wm + 16 + mlane) * 64, s * 32, koff);
#pragma unroll
      for (int t = 0; t < 4; ++t)
        bf[s][t] = *(const v16h*)(Bb + (wn + 16 * t + nlane) * 64 + s * 32 + kb);
    }
    // 16 back-to-back WMMAs
#pragma unroll
    for (int s = 0; s < 2; ++s) {
#pragma unroll
      for (int t = 0; t < 4; ++t) {
        acc[0][t] = wmma_f16(af[s][0], bf[s][t], acc[0][t]);
        acc[1][t] = wmma_f16(af[s][1], bf[s][t], acc[1][t]);
      }
    }
    buf ^= 1;
  }

  const int mloc = (lane & 16) >> 1;  // D tile: rows 0..7 / 8..15 per half-wave
#pragma unroll
  for (int half = 0; half < 2; ++half) {
#pragma unroll
    for (int t = 0; t < 4; ++t) {
#pragma unroll
      for (int r = 0; r < 8; ++r) {
        size_t idx = (size_t)(m0 + wm + 16 * half + mloc + r) * N +
                     (n0 + wn + 16 * t + (lane & 15));
        if (F32OUT) ((float*)out)[idx] = acc[half][t][r];
        else        ((_Float16*)out)[idx] = (_Float16)acc[half][t][r];
      }
    }
  }
}

// ---------------------------------------------------------------------------
// RoPE + repack: Q,K rotated (Q scaled by 1/sqrt(dk)); outputs
//   Qh, Kh : f16 [b,h,s,64]      Vth : f16 [b,h,64,s] (transposed)
// ---------------------------------------------------------------------------
__global__ void rope_pack_kernel(const float* __restrict__ Qf,
                                 const float* __restrict__ Kf,
                                 const float* __restrict__ Vf,
                                 const int* __restrict__ pos,
                                 _Float16* __restrict__ Qh,
                                 _Float16* __restrict__ Kh,
                                 _Float16* __restrict__ Vth) {
  int idx = blockIdx.x * blockDim.x + threadIdx.x;
  const int total = BATCH * SEQ_LEN * D_MODEL;
  if (idx >= total) return;
  int o = idx & (D_MODEL - 1);
  int s = (idx / D_MODEL) & (SEQ_LEN - 1);
  int b = idx / (D_MODEL * SEQ_LEN);
  int h = o >> 6;
  int d = o & 63;
  int t = d >> 1;
  bool odd = d & 1;

  float inv = __powf(10000.0f, -(float)(2 * t) / 64.0f);
  float ang = (float)pos[s] * inv;
  float sn, cs;
  __sincosf(ang, &sn, &cs);

  int base_e = (b * SEQ_LEN + s) * D_MODEL + (o & ~1);
  float qe = Qf[base_e], qo = Qf[base_e + 1];
  float ke = Kf[base_e], ko = Kf[base_e + 1];
  float qv = odd ? (qe * sn + qo * cs) : (qe * cs - qo * sn);
  float kv = odd ? (ke * sn + ko * cs) : (ke * cs - ko * sn);

  const float qscale = 0.125f;  // 1/sqrt(64)
  int bh = b * NUM_HEADS + h;
  Qh[((size_t)bh * SEQ_LEN + s) * DK + d] = (_Float16)(qv * qscale);
  Kh[((size_t)bh * SEQ_LEN + s) * DK + d] = (_Float16)kv;
  Vth[((size_t)bh * DK + d) * SEQ_LEN + s] = (_Float16)Vf[idx];
}

// ---------------------------------------------------------------------------
// Flash attention, causal. 4-wave block per (b, h, 64-query tile); wave w owns
// query rows [q0+16w, q0+16w+16). K/V 32-key blocks shared across all 4 waves
// via TDM-staged double-buffered LDS (4x reuse). V fragments are loaded before
// the softmax so their DS latency hides under the exp/max VALU chain.
// ---------------------------------------------------------------------------
__global__ __launch_bounds__(128)
void attn_kernel(const _Float16* __restrict__ Qh,
                 const _Float16* __restrict__ Kh,
                 const _Float16* __restrict__ Vth,
                 _Float16* __restrict__ Oh) {
  __shared__ _Float16 Pl[4][16 * 32];     // per-wave P staging
  __shared__ _Float16 Kl[2][32 * 64];     // K block: 32 kv rows x 64 d
  __shared__ _Float16 Vl[2][64 * 32];     // V^T block: 64 d rows x 32 kv

  const int tid  = threadIdx.x;
  const int lane = tid & 31;
  const int wid  = tid >> 5;
  const int ntq  = SEQ_LEN / 64;
  const int qt   = blockIdx.x % ntq;
  const int bh   = blockIdx.x / ntq;
  const int b    = bh / NUM_HEADS;
  const int h    = bh % NUM_HEADS;
  const int q0   = qt * 64;               // block's query base
  const int qw   = q0 + wid * 16;         // this wave's query base

  const _Float16* Qb = Qh  + (size_t)bh * SEQ_LEN * DK;
  const _Float16* Kb = Kh  + (size_t)bh * SEQ_LEN * DK;
  const _Float16* Vb = Vth + (size_t)bh * DK * SEQ_LEN;

  const int mlane = lane & 15;
  const int koff  = (lane & 16) >> 1;
  const int nlane = lane & 15;
  const int kb    = lane & 16;
  const int mloc  = (lane & 16) >> 1;

  // Q fragments for this wave's 16 rows (K-dim 64 -> two A frags)
  const _Float16* qrow = Qb + (size_t)(qw + mlane) * DK;
  v16h qa0 = load_a_frag(qrow, 0, koff);
  v16h qa1 = load_a_frag(qrow, 32, koff);

  v8f acc[4] = {};
  float rm[8], rl[8];
#pragma unroll
  for (int r = 0; r < 8; ++r) { rm[r] = -1e30f; rl[r] = 0.0f; }

  const int nblocks = (q0 + 63) / 32 + 1;  // uniform across the block
  int buf = 0;

#if HAVE_TDM
  if (tid < 32) {
    tdm_load_2d(&Kl[0][0], Kb, DK, SEQ_LEN, DK, 32, DK);
    tdm_load_2d(&Vl[0][0], Vb, SEQ_LEN, DK, 32, DK, SEQ_LEN);
  }
#endif
  for (int j = 0; j < nblocks; ++j) {
    const int kv0 = j * 32;
    const bool more = (j + 1 < nblocks);
    __syncthreads();  // readers of buf^1 (block j-1) have drained
#if HAVE_TDM
    if (tid < 32) {
      if (more) {
        tdm_load_2d(&Kl[buf ^ 1][0], Kb + (size_t)(kv0 + 32) * DK,
                    DK, SEQ_LEN, DK, 32, DK);
        tdm_load_2d(&Vl[buf ^ 1][0], Vb + kv0 + 32,
                    SEQ_LEN, DK, 32, DK, SEQ_LEN);
        __builtin_amdgcn_s_wait_tensorcnt(2);
      } else {
        __builtin_amdgcn_s_wait_tensorcnt(0);
      }
    }
#else
    coop_copy(&Kl[buf][0], Kb + (size_t)kv0 * DK, 32, 64, DK, tid, 128);
    coop_copy(&Vl[buf][0], Vb + kv0, 64, 32, SEQ_LEN, tid, 128);
#endif
    __syncthreads();  // publish block j
    const _Float16* Klb = &Kl[buf][0];
    const _Float16* Vlb = &Vl[buf][0];

    // ---- load all K frags (scores) and V frags (PV) up front ----
    v16h kf[2][2];
#pragma unroll
    for (int tt = 0; tt < 2; ++tt) {
      const _Float16* krow = Klb + (16 * tt + nlane) * 64 + kb;
      kf[tt][0] = *(const v16h*)(krow);        // d = kb .. kb+15
      kf[tt][1] = *(const v16h*)(krow + 32);   // d = 32+kb ..
    }
    v16h vf[4];
#pragma unroll
    for (int t2 = 0; t2 < 4; ++t2)
      vf[t2] = *(const v16h*)(Vlb + (16 * t2 + nlane) * 32 + kb);

    // ---- scores: two 16x16 D tiles over kv0..kv0+31 ----
    v8f sc[2];
#pragma unroll
    for (int tt = 0; tt < 2; ++tt) {
      v8f c = {};
      c = wmma_f16(qa0, kf[tt][0], c);
      c = wmma_f16(qa1, kf[tt][1], c);
      sc[tt] = c;
    }

    // ---- causal mask + online softmax (per-wave state) ----
#pragma unroll
    for (int r = 0; r < 8; ++r) {
      int qg  = qw + mloc + r;
      int kvA = kv0 + nlane;
      int kvB = kvA + 16;
      float s0 = (kvA <= qg) ? sc[0][r] : -1e30f;
      float s1 = (kvB <= qg) ? sc[1][r] : -1e30f;

      float v = fmaxf(s0, s1);
      v = fmaxf(v, __shfl_xor(v, 1, 32));
      v = fmaxf(v, __shfl_xor(v, 2, 32));
      v = fmaxf(v, __shfl_xor(v, 4, 32));
      v = fmaxf(v, __shfl_xor(v, 8, 32));

      float mnew  = fmaxf(rm[r], v);
      float alpha = __expf(rm[r] - mnew);
      float p0 = __expf(s0 - mnew);
      float p1 = __expf(s1 - mnew);

      float ps = p0 + p1;
      ps += __shfl_xor(ps, 1, 32);
      ps += __shfl_xor(ps, 2, 32);
      ps += __shfl_xor(ps, 4, 32);
      ps += __shfl_xor(ps, 8, 32);

      rl[r] = rl[r] * alpha + ps;
      rm[r] = mnew;
#pragma unroll
      for (int t2 = 0; t2 < 4; ++t2) acc[t2][r] *= alpha;

      Pl[wid][(mloc + r) * 32 + nlane]      = (_Float16)p0;
      Pl[wid][(mloc + r) * 32 + 16 + nlane] = (_Float16)p1;
    }
    // wave-private P round-trip: same-wave LDS ops complete in order (DScnt)

    // ---- P A-fragment from this wave's LDS slice ----
    const _Float16* prow = &Pl[wid][mlane * 32];
    v8h g0 = *(const v8h*)(prow + 2 * koff);
    v8h g1 = *(const v8h*)(prow + 2 * koff + 8);
    v16h pa = __builtin_shufflevector(g0, g1, 0, 1, 2, 3, 4, 5, 6, 7,
                                              8, 9, 10, 11, 12, 13, 14, 15);

    // ---- O += P @ V ----
#pragma unroll
    for (int t2 = 0; t2 < 4; ++t2)
      acc[t2] = wmma_f16(pa, vf[t2], acc[t2]);

    buf ^= 1;
  }

  // ---- finalize: O / l, write f16 [b, s, h, d] ----
#pragma unroll
  for (int t2 = 0; t2 < 4; ++t2) {
#pragma unroll
    for (int r = 0; r < 8; ++r) {
      int qg = qw + mloc + r;
      float o = acc[t2][r] / rl[r];
      Oh[(((size_t)b * SEQ_LEN + qg) * NUM_HEADS + h) * DK + 16 * t2 + (lane & 15)] =
          (_Float16)o;
    }
  }
}

// ---------------------------------------------------------------------------
// Host-side orchestration
// ---------------------------------------------------------------------------
extern "C" void kernel_launch(void* const* d_in, const int* in_sizes, int n_in,
                              void* d_out, int out_size, void* d_ws, size_t ws_size,
                              hipStream_t stream) {
  (void)in_sizes; (void)n_in; (void)out_size; (void)ws_size;

  const float* x_f  = (const float*)d_in[0];
  const int*   pos  = (const int*)d_in[1];
  const float* qw_f = (const float*)d_in[2];
  const float* kw_f = (const float*)d_in[3];
  const float* vw_f = (const float*)d_in[4];
  const float* ow_f = (const float*)d_in[5];

  const int M = BATCH * SEQ_LEN;    // 4096
  const int K = D_MODEL;            // 1024
  const int N = D_MODEL;            // 1024
  const size_t MD  = (size_t)M * D_MODEL;
  const size_t DD  = (size_t)D_MODEL * D_MODEL;
  const size_t HSD = (size_t)BATCH * NUM_HEADS * SEQ_LEN * DK;

  char* ws = (char*)d_ws;
  _Float16* Xh  = (_Float16*)ws;             ws += MD * 2;
  _Float16* QWh = (_Float16*)ws;             ws += DD * 2;
  _Float16* KWh = (_Float16*)ws;             ws += DD * 2;
  _Float16* VWh = (_Float16*)ws;             ws += DD * 2;
  _Float16* OWh = (_Float16*)ws;             ws += DD * 2;
  float*    Qf  = (float*)ws;                ws += MD * 4;
  float*    Kf  = (float*)ws;                ws += MD * 4;
  float*    Vf  = (float*)ws;                ws += MD * 4;
  _Float16* Qh  = (_Float16*)ws;             ws += HSD * 2;
  _Float16* Kh  = (_Float16*)ws;             ws += HSD * 2;
  _Float16* Vth = (_Float16*)ws;             ws += HSD * 2;
  _Float16* Oh  = (_Float16*)ws;             ws += MD * 2;

  // 1) convert inputs to f16
  {
    int n = (int)MD;
    cvt_f32_to_f16_kernel<<<(n + 255) / 256, 256, 0, stream>>>(x_f, Xh, n);
    n = (int)DD;
    cvt_f32_to_f16_kernel<<<(n + 255) / 256, 256, 0, stream>>>(qw_f, QWh, n);
    cvt_f32_to_f16_kernel<<<(n + 255) / 256, 256, 0, stream>>>(kw_f, KWh, n);
    cvt_f32_to_f16_kernel<<<(n + 255) / 256, 256, 0, stream>>>(vw_f, VWh, n);
    cvt_f32_to_f16_kernel<<<(n + 255) / 256, 256, 0, stream>>>(ow_f, OWh, n);
  }

  // 2) Q/K/V projections (f32 scratch)
  {
    dim3 grid(M / 64, N / 128);
    gemm_awT_kernel<true><<<grid, 128, 0, stream>>>(Xh, QWh, (void*)Qf, M, N, K);
    gemm_awT_kernel<true><<<grid, 128, 0, stream>>>(Xh, KWh, (void*)Kf, M, N, K);
    gemm_awT_kernel<true><<<grid, 128, 0, stream>>>(Xh, VWh, (void*)Vf, M, N, K);
  }

  // 3) RoPE + repack
  {
    int n = (int)MD;
    rope_pack_kernel<<<(n + 255) / 256, 256, 0, stream>>>(Qf, Kf, Vf, pos,
                                                          Qh, Kh, Vth);
  }

  // 4) flash attention (TDM-staged K/V shared by 4 waves)
  {
    int nblk = BATCH * NUM_HEADS * (SEQ_LEN / 64);  // 1024
    attn_kernel<<<nblk, 128, 0, stream>>>(Qh, Kh, Vth, Oh);
  }

  // 5) output projection -> f32 d_out
  {
    dim3 grid(M / 64, N / 128);
    gemm_awT_kernel<true><<<grid, 128, 0, stream>>>(Oh, OWh, d_out, M, N, K);
  }
}
